// rule_attention_1425929142539
// MI455X (gfx1250) — matrix operand order repaired and checked by reading, not compiled, not run on hardware
//
#include <hip/hip_runtime.h>

typedef __attribute__((ext_vector_type(16))) _Float16 v16h;
typedef __attribute__((ext_vector_type(8)))  _Float16 v8h;
typedef __attribute__((ext_vector_type(8)))  float    v8f;

#define CH     64
#define HEADS  4
#define HDIM   256
#define WDIM   256
#define S_TOT  (HDIM * WDIM)
#define SAMP_STRIDE 72   // 64 + 8 pad (144B rows: 16B-aligned, spreads LDS banks)

// ---------------------------------------------------------------------------
// Shared WMMA GEMM: D[64 x 256] = A[64 x 64] (LDS f16) @ B[64 x 256] (LDS f16)
// 8 waves, each wave does 8 of the 64 16x16 tiles; 2 k-steps of 32.
// outbase points at (row0, col0) of the destination; row stride = S_TOT.
// ---------------------------------------------------------------------------
__device__ __forceinline__ void gemm64x256(const _Float16* __restrict__ wlds,
                                           const _Float16* __restrict__ blds,
                                           float* __restrict__ outbase,
                                           const float* __restrict__ bias,
                                           int tid)
{
    const int wave = tid >> 5;
    const int lane = tid & 31;
    const int l15  = lane & 15;
    const int hl   = lane >> 4;

    #pragma unroll
    for (int t = 0; t < 8; ++t) {
        const int tileid = wave * 8 + t;
        const int mt = tileid & 3;    // output-channel tile (0..3)
        const int nt = tileid >> 2;   // pixel tile (0..15)
        v8f acc = {};
        #pragma unroll
        for (int ks = 0; ks < 2; ++ks) {
            // A fragment: lane row m = mt*16 + l15; K chunks at 8*hl and 16+8*hl
            const _Float16* ar = wlds + (mt * 16 + l15) * 64 + ks * 32 + 8 * hl;
            v8h alo = *(const v8h*)(ar);
            v8h ahi = *(const v8h*)(ar + 16);
            v16h a = __builtin_shufflevector(alo, ahi,
                       0,1,2,3,4,5,6,7,8,9,10,11,12,13,14,15);
            // B fragment: lane col n = nt*16 + l15; K chunks at 8*hl and 16+8*hl (+32*ks)
            const _Float16* br = blds + (nt * 16 + l15) * SAMP_STRIDE + ks * 32 + 8 * hl;
            v8h blo = *(const v8h*)(br);
            v8h bhi = *(const v8h*)(br + 16);
            v16h b = __builtin_shufflevector(blo, bhi,
                       0,1,2,3,4,5,6,7,8,9,10,11,12,13,14,15);
            acc = __builtin_amdgcn_wmma_f32_16x16x32_f16(
                      false, a, false, b, (short)0, acc, false, false);
        }
        const int col = nt * 16 + l15;
        #pragma unroll
        for (int j = 0; j < 8; ++j) {
            const int row = mt * 16 + j + 8 * hl;
            float v = acc[j];
            if (bias) v += bias[row];
            outbase[(size_t)row * S_TOT + col] = v;
        }
    }
}

// ---------------------------------------------------------------------------
// Kernel A: offsets -> bilinear sample -> K/Q/V projections (WMMA).
// One block per image row (grid = N*HDIM, 256 threads = 256 pixels).
// ---------------------------------------------------------------------------
__global__ void kern_sample_project(
    const float* __restrict__ x,
    const float* __restrict__ k_off_w, const float* __restrict__ k_off_b,
    const float* __restrict__ k_w,
    const float* __restrict__ q_off_w, const float* __restrict__ q_off_b,
    const float* __restrict__ q_w,
    const float* __restrict__ v_off_w, const float* __restrict__ v_off_b,
    const float* __restrict__ v_w,
    float* __restrict__ Kbuf, float* __restrict__ Qbuf, float* __restrict__ Vbuf)
{
    __shared__ __align__(16) _Float16 wlds[3][64 * 64];
    __shared__ __align__(16) float    offw[6][64];
    __shared__ __align__(16) _Float16 samp[256 * SAMP_STRIDE];

    const int tid = threadIdx.x;
    const int blk = blockIdx.x;
    const int n = blk >> 8;
    const int h = blk & 255;
    const int xc = tid;

    for (int i = tid; i < 64 * 64; i += 256) {
        wlds[0][i] = (_Float16)k_w[i];
        wlds[1][i] = (_Float16)q_w[i];
        wlds[2][i] = (_Float16)v_w[i];
    }
    if (tid < 64) {
        offw[0][tid] = k_off_w[tid];      offw[1][tid] = k_off_w[64 + tid];
        offw[2][tid] = q_off_w[tid];      offw[3][tid] = q_off_w[64 + tid];
        offw[4][tid] = v_off_w[tid];      offw[5][tid] = v_off_w[64 + tid];
    }
    __syncthreads();

    // ---- per-pixel offset dots (6 accumulators over 64 channels) ----
    const float* xpix = x + (size_t)n * CH * S_TOT + h * WDIM + xc;
    float d0 = 0.f, d1 = 0.f, d2 = 0.f, d3 = 0.f, d4 = 0.f, d5 = 0.f;
    for (int c = 0; c < CH; ++c) {
        const float xv = xpix[(size_t)c * S_TOT];
        d0 += offw[0][c] * xv;  d1 += offw[1][c] * xv;
        d2 += offw[2][c] * xv;  d3 += offw[3][c] * xv;
        d4 += offw[4][c] * xv;  d5 += offw[5][c] * xv;
    }
    const float offy[3] = { d0 + k_off_b[0], d2 + q_off_b[0], d4 + v_off_b[0] };
    const float offx[3] = { d1 + k_off_b[1], d3 + q_off_b[1], d5 + v_off_b[1] };
    float* const outs[3] = { Kbuf, Qbuf, Vbuf };

    #pragma unroll
    for (int br = 0; br < 3; ++br) {
        const float py = (float)h + offy[br];
        const float px = (float)xc + offx[br];
        const float y0f = floorf(py), x0f = floorf(px);
        const float wy1 = py - y0f, wy0 = 1.f - wy1;
        const float wx1 = px - x0f, wx0 = 1.f - wx1;
        const int y0 = (int)y0f, x0 = (int)x0f;
        const int y1 = y0 + 1,   x1 = x0 + 1;
        const float my0 = (y0 >= 0 && y0 < HDIM) ? 1.f : 0.f;
        const float my1 = (y1 >= 0 && y1 < HDIM) ? 1.f : 0.f;
        const float mx0 = (x0 >= 0 && x0 < WDIM) ? 1.f : 0.f;
        const float mx1 = (x1 >= 0 && x1 < WDIM) ? 1.f : 0.f;
        const int y0c = min(max(y0, 0), HDIM - 1), y1c = min(max(y1, 0), HDIM - 1);
        const int x0c = min(max(x0, 0), WDIM - 1), x1c = min(max(x1, 0), WDIM - 1);
        const float w00 = wy0 * wx0 * my0 * mx0;
        const float w01 = wy0 * wx1 * my0 * mx1;
        const float w10 = wy1 * wx0 * my1 * mx0;
        const float w11 = wy1 * wx1 * my1 * mx1;
        const int o00 = y0c * WDIM + x0c, o01 = y0c * WDIM + x1c;
        const int o10 = y1c * WDIM + x0c, o11 = y1c * WDIM + x1c;

        const float* xp = x + (size_t)n * CH * S_TOT;
        _Float16* srow = samp + tid * SAMP_STRIDE;
        for (int c = 0; c < CH; ++c) {
            const float* pl = xp + (size_t)c * S_TOT;
            const float val = w00 * pl[o00] + w01 * pl[o01]
                            + w10 * pl[o10] + w11 * pl[o11];
            srow[c] = (_Float16)val;
        }
        __syncthreads();
        float* outbase = outs[br] + (size_t)n * CH * S_TOT + h * WDIM;
        gemm64x256(wlds[br], samp, outbase, nullptr, tid);
        __syncthreads();
    }
}

// ---------------------------------------------------------------------------
// Kernel B: per-(n, key-channel) max over S (key softmax pass 1).
// ---------------------------------------------------------------------------
__global__ void kern_rowmax(const float* __restrict__ Kbuf, float* __restrict__ kmax)
{
    __shared__ float red[256];
    const int row = blockIdx.x;
    const float* r = Kbuf + (size_t)row * S_TOT;
    float m = -3.4e38f;
    for (int i = threadIdx.x; i < S_TOT; i += 256) m = fmaxf(m, r[i]);
    red[threadIdx.x] = m;
    __syncthreads();
    for (int s = 128; s > 0; s >>= 1) {
        if (threadIdx.x < s)
            red[threadIdx.x] = fmaxf(red[threadIdx.x], red[threadIdx.x + s]);
        __syncthreads();
    }
    if (threadIdx.x == 0) kmax[row] = red[0];
}

__global__ void kern_zero(float* __restrict__ p, int nfl)
{
    for (int i = blockIdx.x * 256 + threadIdx.x; i < nfl; i += gridDim.x * 256)
        p[i] = 0.f;
}

// ---------------------------------------------------------------------------
// Kernel C: ctx[k,v] += sum_s exp(K-max)*V  via WMMA; ksum[k] += sum_s exp.
// grid = N * HEADS * 16 chunks; each wave handles 512 s (16 wmma steps).
// ---------------------------------------------------------------------------
__global__ void kern_context(const float* __restrict__ Kbuf,
                             const float* __restrict__ Vbuf,
                             const float* __restrict__ kmax,
                             float* __restrict__ ctx, float* __restrict__ ksum)
{
    const int blk   = blockIdx.x;
    const int chunk = blk & 15;
    const int h     = (blk >> 4) & 3;
    const int n     = blk >> 6;
    const int tid   = threadIdx.x;
    const int wave  = tid >> 5, lane = tid & 31;
    const int l15   = lane & 15, hl = lane >> 4;
    const int sbase0 = chunk * 4096 + wave * 512;

    const float* Kr = Kbuf + ((size_t)n * CH + h * 16 + l15) * S_TOT;
    const float* Vr = Vbuf + ((size_t)n * CH + h * 16 + l15) * S_TOT;
    const float  kmx = kmax[n * CH + h * 16 + l15];

    v8f acc = {};
    float sume = 0.f;
    for (int it = 0; it < 16; ++it) {
        const int sb = sbase0 + it * 32;
        v16h a;
        const float* ka = Kr + sb + 8 * hl;        // elements 0..7
        const float* kb = Kr + sb + 16 + 8 * hl;   // elements 8..15
        #pragma unroll
        for (int e = 0; e < 8; ++e) {
            const float e0 = __expf(ka[e] - kmx);
            const float e1 = __expf(kb[e] - kmx);
            sume += e0 + e1;
            a[e]     = (_Float16)e0;
            a[e + 8] = (_Float16)e1;
        }
        v16h b;
        const float* vb = Vr + sb + 8 * hl;        // k = 8*hl + e (e<8), 16 + 8*hl + (e-8) (e>=8)
        #pragma unroll
        for (int e = 0; e < 16; ++e) b[e] = (_Float16)vb[e + (e >> 3) * 8];
        acc = __builtin_amdgcn_wmma_f32_16x16x32_f16(
                  false, a, false, b, (short)0, acc, false, false);
    }
    #pragma unroll
    for (int j = 0; j < 8; ++j) {
        const int kch = j + 8 * hl;
        atomicAdd(&ctx[(((size_t)n * HEADS + h) * 16 + kch) * 16 + l15], acc[j]);
    }
    sume += __shfl_xor(sume, 16, 32);
    if (lane < 16) atomicAdd(&ksum[n * CH + h * 16 + l15], sume);
}

// ---------------------------------------------------------------------------
// Kernel C2: fuse reprojection with normalized context:
//   W2[o, h*16+k] = sum_v rp_w[o, h*16+v] * ctx[n,h,k,v] / ksum[n, h*16+k]
// ---------------------------------------------------------------------------
__global__ void kern_w2(const float* __restrict__ rp_w, const float* __restrict__ ctx,
                        const float* __restrict__ ksum, _Float16* __restrict__ W2h)
{
    const int n = blockIdx.x;
    for (int idx = threadIdx.x; idx < 4096; idx += 256) {
        const int o = idx >> 6, kg = idx & 63;
        const int h = kg >> 4,  k = kg & 15;
        const float z = ksum[n * CH + kg];
        float s = 0.f;
        #pragma unroll
        for (int v = 0; v < 16; ++v)
            s += rp_w[o * CH + h * 16 + v] * ctx[(((size_t)n * HEADS + h) * 16 + k) * 16 + v];
        W2h[n * 4096 + idx] = (_Float16)(s / z);
    }
}

// ---------------------------------------------------------------------------
// Kernel D: per-pixel query softmax (over 16 channels/head), then
// out = W2 @ qsm + rp_b  (WMMA). One block per image row.
// ---------------------------------------------------------------------------
__global__ void kern_attend(const float* __restrict__ Qbuf,
                            const _Float16* __restrict__ W2h,
                            const float* __restrict__ rp_b,
                            float* __restrict__ out)
{
    __shared__ __align__(16) _Float16 w2l[64 * 64];
    __shared__ __align__(16) _Float16 qsm[256 * SAMP_STRIDE];
    const int tid = threadIdx.x;
    const int blk = blockIdx.x;
    const int n = blk >> 8;
    const int h = blk & 255;

    for (int i = tid; i < 4096; i += 256) w2l[i] = W2h[n * 4096 + i];

    const float* qpix = Qbuf + (size_t)n * CH * S_TOT + h * WDIM + tid;
    _Float16* srow = qsm + tid * SAMP_STRIDE;
    #pragma unroll
    for (int hd = 0; hd < HEADS; ++hd) {
        float q[16];
        float m = -3.4e38f;
        #pragma unroll
        for (int k = 0; k < 16; ++k) {
            q[k] = qpix[(size_t)(hd * 16 + k) * S_TOT];
            m = fmaxf(m, q[k]);
        }
        float sum = 0.f;
        #pragma unroll
        for (int k = 0; k < 16; ++k) { q[k] = __expf(q[k] - m); sum += q[k]; }
        const float inv = 1.f / sum;
        #pragma unroll
        for (int k = 0; k < 16; ++k) srow[hd * 16 + k] = (_Float16)(q[k] * inv);
    }
    __syncthreads();
    float* outbase = out + (size_t)n * CH * S_TOT + h * WDIM;
    gemm64x256(w2l, qsm, outbase, rp_b, tid);
}

// ---------------------------------------------------------------------------
extern "C" void kernel_launch(void* const* d_in, const int* in_sizes, int n_in,
                              void* d_out, int out_size, void* d_ws, size_t ws_size,
                              hipStream_t stream)
{
    const float* x       = (const float*)d_in[0];
    const float* k_off_w = (const float*)d_in[1];
    const float* k_off_b = (const float*)d_in[2];
    const float* k_w     = (const float*)d_in[3];
    const float* q_off_w = (const float*)d_in[4];
    const float* q_off_b = (const float*)d_in[5];
    const float* q_w     = (const float*)d_in[6];
    const float* v_off_w = (const float*)d_in[7];
    const float* v_off_b = (const float*)d_in[8];
    const float* v_w     = (const float*)d_in[9];
    const float* rp_w    = (const float*)d_in[10];
    const float* rp_b    = (const float*)d_in[11];

    const int N = in_sizes[0] / (CH * S_TOT);
    const size_t ncs = (size_t)N * CH * S_TOT;

    float* ws   = (float*)d_ws;
    float* Kbuf = ws;
    float* Qbuf = Kbuf + ncs;
    float* Vbuf = Qbuf + ncs;
    float* kmax = Vbuf + ncs;
    float* ctx  = kmax + (size_t)N * CH;            // N*4*16*16 floats
    float* ksum = ctx  + (size_t)N * HEADS * 256;   // contiguous with ctx
    _Float16* W2h = (_Float16*)(ksum + (size_t)N * CH);

    kern_zero<<<8, 256, 0, stream>>>(ctx, N * HEADS * 256 + N * CH);
    kern_sample_project<<<N * HDIM, 256, 0, stream>>>(
        x, k_off_w, k_off_b, k_w, q_off_w, q_off_b, q_w,
        v_off_w, v_off_b, v_w, Kbuf, Qbuf, Vbuf);
    kern_rowmax<<<N * CH, 256, 0, stream>>>(Kbuf, kmax);
    kern_context<<<N * HEADS * 16, 256, 0, stream>>>(Kbuf, Vbuf, kmax, ctx, ksum);
    kern_w2<<<N, 256, 0, stream>>>(rp_w, ctx, ksum, W2h);
    kern_attend<<<N * HDIM, 256, 0, stream>>>(Qbuf, W2h, rp_b, (float*)d_out);
}
